// PSMGNNModel_75720273428992
// MI455X (gfx1250) — compile-verified, matrix operand order checked by reading, not run on hardware
//
#include <hip/hip_runtime.h>
#include <hip/hip_bf16.h>

typedef __attribute__((ext_vector_type(16))) _Float16 v16h;
typedef __attribute__((ext_vector_type(8)))  float    v8f;
typedef __attribute__((ext_vector_type(4)))  unsigned int v4u;
typedef __attribute__((ext_vector_type(8)))  int      v8i;
typedef __attribute__((ext_vector_type(4)))  int      v4i;

#define NVOX32 32768   // 32^3
#define NVOX16 4096    // 16^3
#define NVOX8  512     // 8^3

struct I16    { int v[16]; };
struct NodeMap{ int cnt[17]; int eid[17][3]; };

typedef __attribute__((address_space(3))) float lds_float;

// ---------------------------------------------------------------------------
// Pairwise 33^3 grouped conv via x-Toeplitz WMMA.
// Block = (edge, z-group-of-4, ytile, xtile), 4 waves; wave w -> z = zg*4+w.
// Per kz: wave0 DMAs the 33x33 f32 weight plane to LDS (TENSOR_LOAD_TO_LDS),
// all threads convert it to a zero-padded f16 Toeplitz source (wpad), each
// wave stages its zero-padded 32(+1 zero)x64 f16 input window; the inner ky
// loop is barrier-free, fully unconditional LDS reads + 2 WMMAs.
// ---------------------------------------------------------------------------
__global__ void k_pairwise(const float* __restrict__ joints,
                           const float* __restrict__ pw,
                           const float* __restrict__ pb,
                           float* __restrict__ out, I16 srcMap) {
  __shared__ _Float16 sl[4 * 33 * 64];       // per-wave slab, row 32 = zeros
  __shared__ float    wplane[33 * 33 + 3];   // TDM target: 33x33 plane (f32)
  __shared__ _Float16 wpad[33 * 96];         // padded taps: [16 zeros|33 taps|47 zeros]
  const int lane = threadIdx.x & 31;
  const int w    = threadIdx.x >> 5;         // wave id 0..3
  int bid = blockIdx.x;
  const int e  = bid & 15; bid >>= 4;
  const int zg = bid & 7;  bid >>= 3;
  const int yt = bid & 1;
  const int xt = bid >> 1;
  const int y0 = yt * 16, x0 = xt * 16;
  const int z  = zg * 4 + w;
  const float* __restrict__ src = joints + (size_t)srcMap.v[e] * NVOX32;

  const int m    = lane & 15;
  const int half = lane >> 4;
  v8f acc = {};

  const unsigned lds_w = (unsigned)(unsigned long long)(lds_float*)wplane;
  _Float16* __restrict__ myslab = &sl[w * (33 * 64)];

  // one-time: zero row 32 of each wave's slab
  for (int c = lane; c < 64; c += 32) myslab[32 * 64 + c] = (_Float16)0.f;

  for (int kz = 0; kz < 33; ++kz) {
    // ---- wave 0: DMA the (e,kz) weight plane (1089 contiguous f32) to LDS ----
    if (threadIdx.x < 32) {
      const unsigned long long ga =
          (unsigned long long)(const void*)(pw + ((size_t)e * 33 + kz) * 1089);
      v4u g0;
      g0[0] = 1u;                                   // count=1, user descriptor
      g0[1] = lds_w;                                // lds_addr
      g0[2] = (unsigned)ga;                         // global_addr[31:0]
      g0[3] = (unsigned)((ga >> 32) & 0x1FFFFFFu) | (2u << 30); // addr[56:32] | type=2
      v8i g1;
      g1[0] = 0x20000;                              // data_size = 4B
      g1[1] = (int)(1089u << 16);                   // tensor_dim0[15:0]
      g1[2] = 0;                                    // tensor_dim0 hi, tensor_dim1 lo
      g1[3] = (int)(1089u << 16);                   // tile_dim0 = 1089
      g1[4] = 0;                                    // tile_dim1 = 0 (1-D)
      g1[5] = 1089;                                 // tensor_dim0_stride lo
      g1[6] = 0;
      g1[7] = 0;
      v4i zz = {0, 0, 0, 0};
      v8i z8 = {0, 0, 0, 0, 0, 0, 0, 0};
      __builtin_amdgcn_tensor_load_to_lds(g0, g1, zz, zz, z8, 0);
      __builtin_amdgcn_s_wait_tensorcnt(0);
    }
    __syncthreads();                                // publish wplane
    // ---- all threads: f32 plane -> zero-padded f16 Toeplitz rows ----
    for (int i = threadIdx.x; i < 33 * 96; i += 128) {
      const int kyy = i / 96, col = i - kyy * 96;
      const float v = (col >= 16 && col < 49) ? wplane[kyy * 33 + (col - 16)] : 0.f;
      wpad[i] = (_Float16)v;
    }
    // ---- each wave stages its zero-padded input window (row = lane) ----
    const int zin = z + kz - 16;
    const bool zok = ((unsigned)zin < 32u);
    if (zok) {
      const float* __restrict__ slice = src + zin * 1024 + lane * 32;
      _Float16* __restrict__ row = &myslab[lane * 64];
      #pragma unroll
      for (int c = 0; c < 64; c += 8)
        for (int q = 0; q < 8; ++q) row[c + q] = (_Float16)0.f;
      const int base = 16 - x0;                     // 16 (xt=0) or 0 (xt=1)
      #pragma unroll 8
      for (int c = 0; c < 32; ++c) row[base + c] = (_Float16)slice[c];
    }
    __syncthreads();
    if (zok) {
      for (int ky = 0; ky < 33; ++ky) {
        const int ybase = y0 + ky - 16;
        if (ybase > 31 || ybase + 15 < 0) continue;         // uniform skip
        const int yin = ybase + m;
        const int rowidx = ((unsigned)yin < 32u) ? yin : 32; // row 32 = zeros
        const _Float16* __restrict__ arow = &myslab[rowidx * 64];
        const _Float16* __restrict__ wr = &wpad[ky * 96 + 16 - m];  // B[k][n]=w[k-n]
        v16h a0, a1, b0, b1;
        #pragma unroll
        for (int r = 0; r < 8; ++r) {
          const int kb = ((r < 4) ? (2 * r) : (2 * r + 8)) + (half ? 8 : 0);
          #pragma unroll
          for (int ee = 0; ee < 2; ++ee) {
            const int k = kb + ee;
            a0[2 * r + ee] = arow[k];
            a1[2 * r + ee] = arow[32 + k];
            b0[2 * r + ee] = wr[k];
            b1[2 * r + ee] = wr[32 + k];
          }
        }
        acc = __builtin_amdgcn_wmma_f32_16x16x32_f16(false, a0, false, b0, (short)0, acc, false, false);
        acc = __builtin_amdgcn_wmma_f32_16x16x32_f16(false, a1, false, b1, (short)0, acc, false, false);
      }
    }
    __syncthreads();
  }
  const float bv = pb[e];
  float* __restrict__ dst = out + (size_t)e * NVOX32 + z * 1024;
  #pragma unroll
  for (int r = 0; r < 8; ++r) {
    const int yy = y0 + r + half * 8;
    dst[yy * 32 + x0 + m] = acc[r] + bv;
  }
}

// ---------------------------------------------------------------------------
// Node product + softmax over the 32^3 volume (one block per joint).
// ---------------------------------------------------------------------------
__global__ void k_node_softmax(const float* __restrict__ x, const float* __restrict__ mv,
                               float* __restrict__ x1, NodeMap nm) {
  __shared__ float red[256];
  const int n = blockIdx.x, tid = threadIdx.x;
  const float* __restrict__ src = x  + (size_t)n * NVOX32;
  float*       __restrict__ dst = x1 + (size_t)n * NVOX32;
  const int cnt = nm.cnt[n];
  if (cnt == 0) { for (int i = tid; i < NVOX32; i += 256) dst[i] = src[i]; return; }
  const float* e0 = mv + (size_t)nm.eid[n][0] * NVOX32;
  const float* e1 = (cnt > 1) ? mv + (size_t)nm.eid[n][1] * NVOX32 : nullptr;
  const float* e2 = (cnt > 2) ? mv + (size_t)nm.eid[n][2] * NVOX32 : nullptr;
  float mx = -3.4e38f;
  for (int i = tid; i < NVOX32; i += 256) {
    float v = src[i] * e0[i];
    if (e1) v *= e1[i];
    if (e2) v *= e2[i];
    dst[i] = v; mx = fmaxf(mx, v);
  }
  red[tid] = mx; __syncthreads();
  for (int s = 128; s > 0; s >>= 1) { if (tid < s) red[tid] = fmaxf(red[tid], red[tid + s]); __syncthreads(); }
  mx = red[0]; __syncthreads();
  float sum = 0.f;
  for (int i = tid; i < NVOX32; i += 256) { float t = expf(dst[i] - mx); dst[i] = t; sum += t; }
  red[tid] = sum; __syncthreads();
  for (int s = 128; s > 0; s >>= 1) { if (tid < s) red[tid] += red[tid + s]; __syncthreads(); }
  const float inv = 1.f / red[0];
  for (int i = tid; i < NVOX32; i += 256) dst[i] *= inv;
}

// ---------------------------------------------------------------------------
// Edge product + per-edge softmax; block 16 copies the root (node 6).
// ---------------------------------------------------------------------------
__global__ void k_edge_softmax(const float* __restrict__ x1, const float* __restrict__ mv,
                               float* __restrict__ x2, I16 childMap) {
  __shared__ float red[256];
  const int b = blockIdx.x, tid = threadIdx.x;
  if (b == 16) {
    const float* s = x1 + (size_t)6 * NVOX32;
    float*       d = x2 + (size_t)6 * NVOX32;
    for (int i = tid; i < NVOX32; i += 256) d[i] = s[i];
    return;
  }
  const float* __restrict__ src = x1 + (size_t)childMap.v[b] * NVOX32;
  const float* __restrict__ m   = mv + (size_t)b * NVOX32;
  float*       __restrict__ dst = x2 + (size_t)childMap.v[b] * NVOX32;
  float mx = -3.4e38f;
  for (int i = tid; i < NVOX32; i += 256) { float v = src[i] * m[i]; dst[i] = v; mx = fmaxf(mx, v); }
  red[tid] = mx; __syncthreads();
  for (int s = 128; s > 0; s >>= 1) { if (tid < s) red[tid] = fmaxf(red[tid], red[tid + s]); __syncthreads(); }
  mx = red[0]; __syncthreads();
  float sum = 0.f;
  for (int i = tid; i < NVOX32; i += 256) { float t = expf(dst[i] - mx); dst[i] = t; sum += t; }
  red[tid] = sum; __syncthreads();
  for (int s = 128; s > 0; s >>= 1) { if (tid < s) red[tid] += red[tid + s]; __syncthreads(); }
  const float inv = 1.f / red[0];
  for (int i = tid; i < NVOX32; i += 256) dst[i] *= inv;
}

// ---------------------------------------------------------------------------
// Generic 3D conv (KS=1 or 3, pad=KS/2) as implicit GEMM on WMMA f16->f32.
// 128-thread block = 4 waves = 4 M-tiles; grid = (Nvox/64, ceil(Cout/16)).
// ---------------------------------------------------------------------------
template <int KS>
__global__ void k_conv_wmma(const float* __restrict__ in, const float* __restrict__ wgt,
                            const float* __restrict__ bias, float* __restrict__ out,
                            int Cin, int Cout, int D, int H, int W) {
  const int lane = threadIdx.x & 31;
  const int m_t  = blockIdx.x * 4 + (threadIdx.x >> 5);
  const int n_t  = blockIdx.y;
  const int Nvox = D * H * W;
  constexpr int KS2 = KS * KS, KS3 = KS * KS * KS;
  const int Ktot = Cin * KS3;
  constexpr int pad = KS >> 1;
  const int m    = m_t * 16 + (lane & 15);
  const int half = lane >> 4;
  const int ncol = n_t * 16 + (lane & 15);
  const int zo = m / (H * W);
  const int rm = m - zo * H * W;
  const int yo = rm / W;
  const int xo = rm - yo * W;
  const float* __restrict__ wrow = wgt + (size_t)ncol * Ktot;  // (Cout, Cin*KS^3)
  v8f acc = {};
  for (int kb = 0; kb < Ktot; kb += 32) {
    if (ncol < Cout) __builtin_prefetch(wrow + kb + 32, 0, 0);
    v16h a, b;
    #pragma unroll
    for (int r = 0; r < 8; ++r) {
      const int kk = ((r < 4) ? (2 * r) : (2 * r + 8)) + (half ? 8 : 0);
      #pragma unroll
      for (int ee = 0; ee < 2; ++ee) {
        const int k = kb + kk + ee;
        float av = 0.f, bv = 0.f;
        if (k < Ktot) {
          const int ci  = k / KS3;
          const int rem = k - ci * KS3;
          const int kz  = rem / KS2;
          const int r2  = rem - kz * KS2;
          const int kyy = r2 / KS;
          const int kxx = r2 - kyy * KS;
          const int zi = zo + kz - pad, yi = yo + kyy - pad, xi = xo + kxx - pad;
          if ((unsigned)zi < (unsigned)D && (unsigned)yi < (unsigned)H && (unsigned)xi < (unsigned)W)
            av = in[((size_t)ci * D + zi) * H * W + yi * W + xi];
          if (ncol < Cout) bv = wrow[k];
        }
        a[2 * r + ee] = (_Float16)av;
        b[2 * r + ee] = (_Float16)bv;
      }
    }
    acc = __builtin_amdgcn_wmma_f32_16x16x32_f16(false, a, false, b, (short)0, acc, false, false);
  }
  if (ncol < Cout) {
    const float bb = bias[ncol];
    #pragma unroll
    for (int r = 0; r < 8; ++r) {
      const int mm = m_t * 16 + r + half * 8;
      out[(size_t)ncol * Nvox + mm] = acc[r] + bb;
    }
  }
}

// ---------------------------------------------------------------------------
// BatchNorm (training-mode batch stats) helpers + elementwise ops.
// ---------------------------------------------------------------------------
__global__ void k_bn_stats(const float* __restrict__ x, float* __restrict__ mean,
                           float* __restrict__ var, int Nvox) {
  __shared__ float s1[256], s2[256];
  const int c = blockIdx.x, tid = threadIdx.x;
  const float* p = x + (size_t)c * Nvox;
  float a = 0.f, b = 0.f;
  for (int i = tid; i < Nvox; i += 256) { float v = p[i]; a += v; b += v * v; }
  s1[tid] = a; s2[tid] = b; __syncthreads();
  for (int s = 128; s > 0; s >>= 1) {
    if (tid < s) { s1[tid] += s1[tid + s]; s2[tid] += s2[tid + s]; }
    __syncthreads();
  }
  if (tid == 0) {
    const float mu = s1[0] / (float)Nvox;
    mean[c] = mu;
    var[c]  = s2[0] / (float)Nvox - mu * mu;
  }
}

__global__ void k_bn_apply(const float* __restrict__ x, const float* __restrict__ mean,
                           const float* __restrict__ var, const float* __restrict__ g,
                           const float* __restrict__ be, const float* __restrict__ add,
                           int relu, float* __restrict__ out, int Nvox, int total) {
  const int i = blockIdx.x * 256 + threadIdx.x;
  if (i >= total) return;
  const int c = i / Nvox;
  float y = (x[i] - mean[c]) * rsqrtf(var[c] + 1e-5f) * g[c] + be[c];
  if (add) y += add[i];
  if (relu) y = fmaxf(y, 0.f);
  out[i] = y;
}

__global__ void k_pool(const float* __restrict__ in, float* __restrict__ out,
                       int C, int oD, int oH, int oW) {
  const int i = blockIdx.x * 256 + threadIdx.x;
  const int total = C * oD * oH * oW;
  if (i >= total) return;
  int t = i;
  const int x = t % oW; t /= oW;
  const int y = t % oH; t /= oH;
  const int z = t % oD; const int c = t / oD;
  const int iH = oH * 2, iW = oW * 2, iD = oD * 2;
  const float* p = in + ((size_t)c * iD + z * 2) * iH * iW;
  float mx = -3.4e38f;
  for (int dz = 0; dz < 2; ++dz)
    for (int dy = 0; dy < 2; ++dy)
      for (int dx = 0; dx < 2; ++dx)
        mx = fmaxf(mx, p[dz * iH * iW + (y * 2 + dy) * iW + (x * 2 + dx)]);
  out[i] = mx;
}

// ConvTranspose3d(k=2, s=2): each output voxel maps to exactly one input voxel.
__global__ void k_upsample(const float* __restrict__ in, const float* __restrict__ wt,
                           const float* __restrict__ bias, float* __restrict__ out,
                           int Cin, int Cout, int iD, int iH, int iW) {
  const int oD = iD * 2, oH = iH * 2, oW = iW * 2;
  const int i = blockIdx.x * 256 + threadIdx.x;
  const int total = Cout * oD * oH * oW;
  if (i >= total) return;
  int t = i;
  const int x = t % oW; t /= oW;
  const int y = t % oH; t /= oH;
  const int z = t % oD; const int co = t / oD;
  const int ktap = (z & 1) * 4 + (y & 1) * 2 + (x & 1);
  const size_t inoff = (size_t)(z >> 1) * iH * iW + (size_t)(y >> 1) * iW + (x >> 1);
  float s = bias[co];
  for (int ci = 0; ci < Cin; ++ci)
    s += in[(size_t)ci * iD * iH * iW + inoff] * wt[((size_t)ci * Cout + co) * 8 + ktap];
  out[i] = s;
}

__global__ void k_add(const float* __restrict__ a, const float* __restrict__ b,
                      float* __restrict__ o, int n) {
  const int i = blockIdx.x * 256 + threadIdx.x;
  if (i < n) o[i] = a[i] + b[i];
}

// ---------------------------------------------------------------------------
// Host orchestration.
// ---------------------------------------------------------------------------
struct ResP {
  const float *w1, *b1, *g1, *be1, *w2, *b2, *g2, *be2, *ws, *bs, *gs, *bes;
  bool has;
};

extern "C" void kernel_launch(void* const* d_in, const int* in_sizes, int n_in,
                              void* d_out, int out_size, void* d_ws, size_t ws_size,
                              hipStream_t stream) {
  auto F = [&](int i) { return (const float*)d_in[i]; };
  const float* X  = F(0);
  const float* PW = F(1);
  const float* PB = F(2);

  auto getres = [&](int base, bool hasws) {
    ResP p;
    p.w1 = F(base);     p.b1 = F(base + 1); p.g1 = F(base + 2); p.be1 = F(base + 3);
    p.w2 = F(base + 4); p.b2 = F(base + 5); p.g2 = F(base + 6); p.be2 = F(base + 7);
    p.has = hasws;
    if (hasws) { p.ws = F(base + 8); p.bs = F(base + 9); p.gs = F(base + 10); p.bes = F(base + 11); }
    else       { p.ws = p.bs = p.gs = p.bes = nullptr; }
    return p;
  };
  ResP skip_res1 = getres(3,  false);
  ResP enc_res1  = getres(11, true);
  ResP skip_res2 = getres(23, false);
  ResP enc_res2  = getres(31, true);
  ResP dec_res2  = getres(43, false);
  ResP dec_res1  = getres(51, false);
  const float *up2_wt = F(59), *up2_b = F(60), *up2_g = F(61), *up2_be = F(62);
  const float *up1_wt = F(63), *up1_b = F(64), *up1_g = F(65), *up1_be = F(66);

  float* wsf = (float*)d_ws;
  size_t off = 0;
  auto A = [&](size_t n) { float* p = wsf + off; off += n; return p; };
  float* X1   = A(17u * NVOX32);
  float* X2   = A(17u * NVOX32);
  float* MV   = A(16u * NVOX32);
  float* S1   = A(17u * NVOX32);
  float* S2   = A(32u * NVOX16);
  float* P16  = A(17u * NVOX16);
  float* P8   = A(32u * NVOX8);
  float* H16  = A(32u * NVOX16);
  float* H8   = A(64u * NVOX8);
  float* H8B  = A(64u * NVOX8);
  float* H16B = A(32u * NVOX16);
  float* U16  = A(32u * NVOX16);
  float* T0   = A(32u * NVOX32);
  float* T1   = A(32u * NVOX32);
  float* T2   = A(32u * NVOX32);
  float* MEAN = A(64);
  float* VAR  = A(64);
  (void)ws_size; (void)n_in; (void)in_sizes; (void)out_size;

  I16 childMap  = {{1, 2, 4, 5, 0, 3, 7, 8, 9, 11, 14, 10, 12, 13, 15, 16}};
  I16 parentMap = {{0, 1, 3, 4, 6, 6, 6, 7, 8, 8, 8, 9, 11, 12, 14, 15}};
  NodeMap nm = {};
  for (int n = 0; n < 17; ++n) nm.cnt[n] = 0;
  for (int e = 0; e < 16; ++e) {
    const int p = parentMap.v[e];
    nm.eid[p][nm.cnt[p]++] = e;
  }

  // ---- Stage 1: pairwise message passing ----
  k_pairwise<<<dim3(512), dim3(128), 0, stream>>>(X, PW, PB, MV, childMap);
  k_node_softmax<<<dim3(17), dim3(256), 0, stream>>>(X, MV, X1, nm);
  k_pairwise<<<dim3(512), dim3(128), 0, stream>>>(X1, PW, PB, MV, parentMap);
  k_edge_softmax<<<dim3(17), dim3(256), 0, stream>>>(X1, MV, X2, childMap);

  // ---- Stage 2: encoder / decoder CNN ----
  auto resblock = [&](const float* in, const ResP& p, int Cin, int Cout,
                      int D, int H, int W, float* out) {
    const int Nvox = D * H * W;
    const dim3 g1(Nvox / 64, (Cout + 15) / 16);
    const int ew = (Cout * Nvox + 255) / 256;
    k_conv_wmma<3><<<g1, 128, 0, stream>>>(in, p.w1, p.b1, T0, Cin, Cout, D, H, W);
    k_bn_stats<<<Cout, 256, 0, stream>>>(T0, MEAN, VAR, Nvox);
    k_bn_apply<<<ew, 256, 0, stream>>>(T0, MEAN, VAR, p.g1, p.be1, nullptr, 1, T1, Nvox, Cout * Nvox);
    const float* skip = in;
    if (p.has) {
      k_conv_wmma<1><<<g1, 128, 0, stream>>>(in, p.ws, p.bs, T2, Cin, Cout, D, H, W);
      k_bn_stats<<<Cout, 256, 0, stream>>>(T2, MEAN, VAR, Nvox);
      k_bn_apply<<<ew, 256, 0, stream>>>(T2, MEAN, VAR, p.gs, p.bes, nullptr, 0, T2, Nvox, Cout * Nvox);
      skip = T2;
    }
    k_conv_wmma<3><<<g1, 128, 0, stream>>>(T1, p.w2, p.b2, T0, Cout, Cout, D, H, W);
    k_bn_stats<<<Cout, 256, 0, stream>>>(T0, MEAN, VAR, Nvox);
    k_bn_apply<<<ew, 256, 0, stream>>>(T0, MEAN, VAR, p.g2, p.be2, skip, 1, out, Nvox, Cout * Nvox);
  };

  resblock(X2, skip_res1, 17, 17, 32, 32, 32, S1);                     // s1
  k_pool<<<(17 * NVOX16 + 255) / 256, 256, 0, stream>>>(X2, P16, 17, 16, 16, 16);
  resblock(P16, enc_res1, 17, 32, 16, 16, 16, H16);
  resblock(H16, skip_res2, 32, 32, 16, 16, 16, S2);                    // s2
  k_pool<<<(32 * NVOX8 + 255) / 256, 256, 0, stream>>>(H16, P8, 32, 8, 8, 8);
  resblock(P8, enc_res2, 32, 64, 8, 8, 8, H8);
  resblock(H8, dec_res2, 64, 64, 8, 8, 8, H8B);

  // up2: 8^3 x64 -> 16^3 x32, BN+ReLU, then + s2
  k_upsample<<<(32 * NVOX16 + 255) / 256, 256, 0, stream>>>(H8B, up2_wt, up2_b, T0, 64, 32, 8, 8, 8);
  k_bn_stats<<<32, 256, 0, stream>>>(T0, MEAN, VAR, NVOX16);
  k_bn_apply<<<(32 * NVOX16 + 255) / 256, 256, 0, stream>>>(T0, MEAN, VAR, up2_g, up2_be,
                                                            nullptr, 1, T1, NVOX16, 32 * NVOX16);
  k_add<<<(32 * NVOX16 + 255) / 256, 256, 0, stream>>>(T1, S2, U16, 32 * NVOX16);

  resblock(U16, dec_res1, 32, 32, 16, 16, 16, H16B);

  // up1: 16^3 x32 -> 32^3 x17, BN+ReLU, then + s1 -> d_out
  k_upsample<<<(17 * NVOX32 + 255) / 256, 256, 0, stream>>>(H16B, up1_wt, up1_b, T0, 32, 17, 16, 16, 16);
  k_bn_stats<<<17, 256, 0, stream>>>(T0, MEAN, VAR, NVOX32);
  k_bn_apply<<<(17 * NVOX32 + 255) / 256, 256, 0, stream>>>(T0, MEAN, VAR, up1_g, up1_be,
                                                            nullptr, 1, T1, NVOX32, 17 * NVOX32);
  k_add<<<(17 * NVOX32 + 255) / 256, 256, 0, stream>>>(T1, S1, (float*)d_out, 17 * NVOX32);
}
